// GraphConvolution_84911503442515
// MI455X (gfx1250) — matrix operand order repaired and checked by reading, not compiled, not run on hardware
//
#include <hip/hip_runtime.h>
#include <stdint.h>
#include <stddef.h>

// ---------------------------------------------------------------------------
// GraphConvolution forward for MI455X (gfx1250, wave32, WMMA).
//   out = d_inv ⊙ (A @ h + delta ⊙ h),  h = leaky_relu(x @ W^T)
// A stays f32 in HBM (134MB, L2-resident at 192MB), converted to bf16 tiles in
// LDS; matmuls on v_wmma_f32_16x16x32_bf16 (f32 accumulate).  h is stored
// TRANSPOSED (hT[b][e][n], bf16) so the main kernel's B-tiles are staged with
// GLOBAL_LOAD_ASYNC_TO_LDS_B128 (ASYNCcnt) straight into the WMMA-friendly
// [e][k] layout.  Main loop is double-buffered: stage chunk i+1 while
// computing chunk i, one barrier per chunk.
// ---------------------------------------------------------------------------

typedef __attribute__((ext_vector_type(16))) __bf16 v16bf;
typedef __attribute__((ext_vector_type(8)))  float  v8f;
typedef __attribute__((ext_vector_type(4)))  float  f32x4;

#define NEG_SLOPE 0.01f
#define BB   8
#define NN   2048
#define DD   128

__device__ __forceinline__ v8f wmma_bf16(v16bf a, v16bf b, v8f c) {
  // (neg_a, A, neg_b, B, c_mod, C, reuse_a, reuse_b)
  return __builtin_amdgcn_wmma_f32_16x16x32_bf16(false, a, false, b,
                                                 (short)0, c, false, false);
}

// B-fragment: 16 contiguous bf16 per lane (K contiguous), 2 x ds_load_b128.
__device__ __forceinline__ v16bf frag_contig(const __bf16* p) {
  union { f32x4 q[2]; v16bf v; } u;
  u.q[0] = *(const f32x4*)(p);
  u.q[1] = *(const f32x4*)(p + 8);
  return u.v;
}

// A-fragment per ISA layout: elems 0..7 -> K = s+0..7, elems 8..15 -> 16+s+0..7
// (s = 0 for lanes 0-15, s = 8 for lanes 16-31)
__device__ __forceinline__ v16bf frag_a(const __bf16* rowp, int s) {
  union { f32x4 q[2]; v16bf v; } u;
  u.q[0] = *(const f32x4*)(rowp + s);
  u.q[1] = *(const f32x4*)(rowp + 16 + s);
  return u.v;
}

__device__ __forceinline__ void wait_asynccnt0() {
#if __has_builtin(__builtin_amdgcn_s_wait_asynccnt)
  __builtin_amdgcn_s_wait_asynccnt(0);
#else
  asm volatile("s_wait_asynccnt 0x0" ::: "memory");
#endif
}

// Async global->LDS copy of 16 bytes per lane (GV addressing).
// lds byte offset = low 32 bits of the generic shared-pointer.
__device__ __forceinline__ void async_copy_b128(const void* gptr, void* lptr) {
  uint32_t lds_off = (uint32_t)(uintptr_t)lptr;
  asm volatile("global_load_async_to_lds_b128 %0, %1, off"
               :: "v"(lds_off), "v"(gptr) : "memory");
}

// ---------------------------------------------------------------------------
// Kernel 1: column sums of A (for the symmetrization mask).  Coalesced.
// ---------------------------------------------------------------------------
__global__ void __launch_bounds__(256) k_colsum(const float* __restrict__ A,
                                                float* __restrict__ colsum) {
  const int b = blockIdx.y;
  const int n = blockIdx.x * 256 + threadIdx.x;
  const float* base = A + (size_t)b * NN * NN + n;
  float acc = 0.f;
  for (int m = 0; m < NN; ++m) acc += base[(size_t)m * NN];
  colsum[b * NN + n] = acc;
}

// ---------------------------------------------------------------------------
// Kernel 2: row sums + mask + diagonal patch -> d_inv, delta per (b,n).
// ---------------------------------------------------------------------------
__global__ void __launch_bounds__(256) k_rowstats(const float* __restrict__ A,
                                                  const float* __restrict__ colsum,
                                                  float* __restrict__ dinv,
                                                  float* __restrict__ delta) {
  __shared__ float sdata[256];
  const int b = blockIdx.y, n = blockIdx.x, t = threadIdx.x;
  const float* row = A + ((size_t)b * NN + n) * NN;
  float acc = 0.f;
  for (int m = t; m < NN; m += 256) acc += row[m];
  sdata[t] = acc;
  __syncthreads();
  for (int s = 128; s > 0; s >>= 1) {
    if (t < s) sdata[t] += sdata[t + s];
    __syncthreads();
  }
  if (t == 0) {
    const float rs   = sdata[0];
    const float diag = row[n];
    const float cs   = colsum[b * NN + n];
    const bool  mask = (rs + cs) != 0.f;        // row of (A + A^T) nonzero?
    const float nd   = mask ? 1.f : diag;       // patched diagonal
    const float d    = rs - diag + nd;          // row sum of patched A
    dinv[b * NN + n]  = (d != 0.f) ? (1.f / d) : 0.f;
    delta[b * NN + n] = nd - diag;              // diagonal correction term
  }
}

// ---------------------------------------------------------------------------
// Kernel 3: h = leaky_relu(x @ W^T), written TRANSPOSED: hT[b][e][n] (bf16).
// 128x128 tile per block; output transposed through LDS for coalesced stores.
// ---------------------------------------------------------------------------
__global__ void __launch_bounds__(256) k_h(const float* __restrict__ x,
                                           const float* __restrict__ W,
                                           __bf16* __restrict__ hT) {
  __shared__ __align__(16) __bf16 lds_x[128 * 128];   // [m][d], later [e][m]
  __shared__ __align__(16) __bf16 lds_w[128 * 128];   // [e][d]
  const int t    = threadIdx.x;
  const int row0 = blockIdx.x * 128;

  // Stage W and the x tile as bf16 (coalesced f32x4 reads).
  #pragma unroll
  for (int j = 0; j < 16; ++j) {
    const int idx = t + j * 256;                       // 0..4095 f32x4 units
    f32x4 wv = *(const f32x4*)(W + (size_t)idx * 4);
    f32x4 xv = *(const f32x4*)(x + (size_t)row0 * DD + (size_t)idx * 4);
    #pragma unroll
    for (int c = 0; c < 4; ++c) {
      lds_w[idx * 4 + c] = (__bf16)wv[c];
      lds_x[idx * 4 + c] = (__bf16)xv[c];
    }
  }
  __syncthreads();

  const int lane = t & 31, w = t >> 5;
  const int lhi = lane >> 4, ln = lane & 15;
  const int s = lhi * 8, s2 = lhi * 16;

  v8f acc[8];
  #pragma unroll
  for (int j = 0; j < 8; ++j) acc[j] = v8f{0.f, 0.f, 0.f, 0.f, 0.f, 0.f, 0.f, 0.f};

  const __bf16* arow = lds_x + (16 * w + ln) * 128;    // wave's 16-row strip
  #pragma unroll
  for (int kc = 0; kc < 4; ++kc) {                     // K = 128 in 4 chunks
    v16bf a = frag_a(arow + kc * 32, s);
    #pragma unroll
    for (int j = 0; j < 8; ++j) {
      v16bf bf = frag_a(lds_w + (16 * j + ln) * 128 + kc * 32, s);
      acc[j] = wmma_bf16(a, bf, acc[j]);
    }
  }

  __syncthreads();   // everyone done reading lds_x; reuse it as [e][m] tile

  // Leaky-ReLU + transpose into LDS.  C/D elem r -> M = 16w + r + 8*lhi, N = ln.
  #pragma unroll
  for (int j = 0; j < 8; ++j) {
    #pragma unroll
    for (int r = 0; r < 8; ++r) {
      const int ml = 16 * w + r + lhi * 8;
      const int e  = 16 * j + ln;
      float v = acc[j][r];
      v = (v > 0.f) ? v : NEG_SLOPE * v;
      lds_x[e * 128 + ml] = (__bf16)v;
    }
  }
  __syncthreads();

  // Coalesced transposed store: hT[(b*DD+e)*NN + n0 + q], 16B chunks.
  const int b  = row0 >> 11;          // row0 / NN
  const int n0 = row0 & (NN - 1);
  #pragma unroll
  for (int j = 0; j < 8; ++j) {
    const int c = t + j * 256;        // 0..2047 chunks of 8 bf16
    const int e = c >> 4;
    const int q = (c & 15) * 8;
    *(f32x4*)(hT + ((size_t)b * DD + e) * NN + n0 + q) =
        *(const f32x4*)(lds_x + e * 128 + q);
  }
}

// ---------------------------------------------------------------------------
// Kernel 4: out = d_inv ⊙ (A @ h + delta ⊙ h).
// 64-row x 128-col tile per block (256 blocks), K-chunk 64, double-buffered:
//   A tile:  global f32 -> VALU cvt -> LDS bf16 [m][k]
//   h tile:  GLOBAL_LOAD_ASYNC_TO_LDS_B128 straight into [e][k] (no VGPRs)
// 8 waves: wave = (row strip sr = w&3, column half ch = w>>2), 4 accumulators.
// ---------------------------------------------------------------------------
__global__ void __launch_bounds__(256) k_main(const float* __restrict__ A,
                                              const __bf16* __restrict__ hT,
                                              const float* __restrict__ dinv,
                                              const float* __restrict__ delta,
                                              float* __restrict__ out) {
  __shared__ __align__(16) __bf16 lds_a[2][64 * 64];    // [m][k] 8KB  x2
  __shared__ __align__(16) __bf16 lds_h[2][128 * 64];   // [e][k] 16KB x2
  const int t  = threadIdx.x;
  const int b  = blockIdx.y;
  const int r0 = blockIdx.x * 64;
  const float*  Ab  = A  + (size_t)b * NN * NN;
  const __bf16* hTb = hT + (size_t)b * DD * NN;

  const int lane = t & 31, w = t >> 5;
  const int lhi = lane >> 4, ln = lane & 15;
  const int s = lhi * 8, s2 = lhi * 16;
  const int sr = w & 3;        // row strip: rows 16*sr .. 16*sr+15
  const int ch = w >> 2;       // column half: cols 64*ch .. 64*ch+63

  v8f acc[4];
  #pragma unroll
  for (int j = 0; j < 4; ++j) acc[j] = v8f{0.f, 0.f, 0.f, 0.f, 0.f, 0.f, 0.f, 0.f};

  // Stage one 64-K chunk into buffer `buf`.
  auto stage = [&](int buf, int k0) {
    // A: 64x64 f32 -> bf16 (1024 f32x4 / 256 thr = 4 each, coalesced).
    #pragma unroll
    for (int j = 0; j < 4; ++j) {
      const int idx = t + j * 256;
      const int row = idx >> 4;
      const int c4  = (idx & 15) * 4;
      const float* src = Ab + (size_t)(r0 + row) * NN + k0 + c4;
      if (k0 + 64 < NN) __builtin_prefetch(src + 64, 0, 1);   // next K chunk
      f32x4 v = *(const f32x4*)src;
      #pragma unroll
      for (int c = 0; c < 4; ++c) lds_a[buf][row * 64 + c4 + c] = (__bf16)v[c];
    }
    // hT tile: 128 rows(e) x 64 k bf16 = 1024 x 16B chunks, async to LDS.
    #pragma unroll
    for (int j = 0; j < 4; ++j) {
      const int c = t + j * 256;
      const int e = c >> 3;
      const int q = (c & 7) * 8;
      async_copy_b128(hTb + (size_t)e * NN + k0 + q,
                      &lds_h[buf][e * 64 + q]);
    }
  };

  stage(0, 0);
  wait_asynccnt0();
  __syncthreads();

  int buf = 0;
  for (int k0 = 0; k0 < NN; k0 += 64) {
    if (k0 + 64 < NN) stage(buf ^ 1, k0 + 64);   // overlap with compute below

    const __bf16* arow = lds_a[buf] + (16 * sr + ln) * 64;
    #pragma unroll
    for (int kc = 0; kc < 2; ++kc) {             // two K=32 sub-chunks
      v16bf a = frag_a(arow + kc * 32, s);
      #pragma unroll
      for (int j = 0; j < 4; ++j) {
        const int e = 64 * ch + 16 * j + ln;
        v16bf bf = frag_a(lds_h[buf] + e * 64 + kc * 32, s);
        acc[j] = wmma_bf16(a, bf, acc[j]);
      }
    }

    wait_asynccnt0();      // my async stage of buf^1 done
    __syncthreads();       // everyone's compute of buf done, stage visible
    buf ^= 1;
  }

  // Epilogue: out[gm,:] = dinv[gm] * (acc + delta[gm] * h[gm,:]).
  #pragma unroll
  for (int r = 0; r < 8; ++r) {
    const int gm = r0 + 16 * sr + r + lhi * 8;
    const float di = dinv[b * NN + gm];
    const float de = delta[b * NN + gm];
    #pragma unroll
    for (int j = 0; j < 4; ++j) {
      const int gn = 64 * ch + 16 * j + ln;
      const float hv = (float)hTb[(size_t)gn * NN + gm];
      out[((size_t)b * NN + gm) * DD + gn] = di * (acc[j][r] + de * hv);
    }
  }
}

// ---------------------------------------------------------------------------
// Host launcher.
// Inputs (setup_inputs order): d_in[0]=x (8,2048,128) f32, d_in[1]=A (8,2048,2048)
// f32, d_in[2]=W (128,128) f32.  Output: (8,2048,128) f32.
// Workspace: dinv[16384] f32 | delta[16384] f32 | colsum[16384] f32 |
//            hT[8*128*2048] bf16   (~4.4 MB total)
// ---------------------------------------------------------------------------
extern "C" void kernel_launch(void* const* d_in, const int* in_sizes, int n_in,
                              void* d_out, int out_size, void* d_ws, size_t ws_size,
                              hipStream_t stream) {
  const float* x = (const float*)d_in[0];
  const float* A = (const float*)d_in[1];
  const float* W = (const float*)d_in[2];
  float* out = (float*)d_out;

  float*  dinv   = (float*)d_ws;
  float*  delta  = dinv + BB * NN;
  float*  colsum = delta + BB * NN;
  __bf16* hT     = (__bf16*)(colsum + BB * NN);

  // 1) column sums (mask input)
  k_colsum<<<dim3(NN / 256, BB), 256, 0, stream>>>(A, colsum);
  // 2) row stats -> d_inv, delta
  k_rowstats<<<dim3(NN, BB), 256, 0, stream>>>(A, colsum, dinv, delta);
  // 3) hT = leaky_relu(x @ W^T)^T in bf16
  k_h<<<dim3((BB * NN) / 128), 256, 0, stream>>>(x, W, hT);
  // 4) out = dinv * (A @ h + delta * h)
  k_main<<<dim3(NN / 64, BB), 256, 0, stream>>>(A, hT, dinv, delta, out);
}
